// TripletLoss_OIM_90185723281456
// MI455X (gfx1250) — matrix operand order repaired and checked by reading, not compiled, not run on hardware
//
#include <hip/hip_runtime.h>
#include <hip/hip_bf16.h>
#include <cfloat>

// ---------------------------------------------------------------------------
// Batch-hard triplet loss (OIM), fused for MI455X / gfx1250, wave32 + WMMA.
//   B = 4096 rows, D = 1024 features, GEMM M=N=B, K=D.
// Pipeline:
//   prep_feat    : feat f32 -> bf16 (RNE) + sq_a (f32, from original f32 data)
//   prep_gather  : lut[id]  -> bf16 (RNE) + sq_b
//   triplet_main : fused bf16 WMMA GEMM + dist + masked row max/min
//                  grid = (B/64) x NCHUNK; 128KB dynamic-LDS A tile (64 rows)
//                  -> halves L2 traffic for the streamed B operand vs 32-row tiles
//   finalize     : combine per-chunk partials, hinge, write loss
// ---------------------------------------------------------------------------

typedef __attribute__((ext_vector_type(16))) __bf16       v16bf;
typedef __attribute__((ext_vector_type(8)))  float        v8f;
typedef __attribute__((ext_vector_type(4)))  float        v4f;
typedef __attribute__((ext_vector_type(4)))  unsigned int v4u;
typedef __attribute__((ext_vector_type(2)))  unsigned int v2u;

#define TL_MARGIN 0.3f
#define TL_BIG    1e5f
#define NCHUNK    4          // N split across blockIdx.y for parallelism

struct alignas(32) FragBits { v4u lo, hi; };   // 32B = one bf16 fragment half-pair

__device__ __forceinline__ unsigned int f32_to_bf16_bits(float x) {
    unsigned int u = __float_as_uint(x);
    u += 0x7FFFu + ((u >> 16) & 1u);   // round-to-nearest-even
    return u >> 16;
}

// -------- shared body: convert one f32 row to bf16, accumulate sum(x*x) ----
__device__ __forceinline__ void convert_row_and_norm(const float* __restrict__ src,
                                                     unsigned short* __restrict__ dst,
                                                     float* __restrict__ sqOut,
                                                     int row, int D) {
    const int tid = threadIdx.x;
    float acc = 0.0f;
    for (int k = tid * 4; k < D; k += 256 * 4) {
        v4f x = *(const v4f*)(src + k);
        acc += x[0] * x[0] + x[1] * x[1] + x[2] * x[2] + x[3] * x[3];
        unsigned int p0 = f32_to_bf16_bits(x[0]) | (f32_to_bf16_bits(x[1]) << 16);
        unsigned int p1 = f32_to_bf16_bits(x[2]) | (f32_to_bf16_bits(x[3]) << 16);
        v2u p = {p0, p1};
        *(v2u*)(dst + k) = p;
    }
    #pragma unroll
    for (int m = 16; m >= 1; m >>= 1) acc += __shfl_xor(acc, m, 32);
    __shared__ float wsum[8];
    if ((tid & 31) == 0) wsum[tid >> 5] = acc;
    __syncthreads();
    if (tid == 0) {
        float s = 0.0f;
        #pragma unroll
        for (int i = 0; i < 8; ++i) s += wsum[i];
        sqOut[row] = s;
    }
}

__global__ __launch_bounds__(256)
void prep_feat(const float* __restrict__ feat, unsigned short* __restrict__ outBf,
               float* __restrict__ sq, int D) {
    const int row = blockIdx.x;
    convert_row_and_norm(feat + (size_t)row * D, outBf + (size_t)row * D, sq, row, D);
}

__global__ __launch_bounds__(256)
void prep_gather(const float* __restrict__ lut, const int* __restrict__ id,
                 unsigned short* __restrict__ outBf, float* __restrict__ sq, int D) {
    const int row = blockIdx.x;
    const int src_row = id[row];
    convert_row_and_norm(lut + (size_t)src_row * D, outBf + (size_t)row * D, sq, row, D);
}

// ---------------------------------------------------------------------------
// Main fused kernel.
//   Block: 256 threads = 8 waves, arranged 4 (M) x 2 (N).
//   Block tile: 64 rows of A, full K resident in 128KB dynamic LDS
//               (fragment-permuted: conflict-free 32B/lane reads).
//   Wave tile : 16 rows x 64 cols -> 4 independent WMMA accumulator chains,
//               each A fragment (2x ds_load_b128) feeds 4 WMMAs.
//   N loop    : covers [nStart, nStart + Bn/NCHUNK) in steps of 128 cols.
// ---------------------------------------------------------------------------
__global__ __launch_bounds__(256)
void triplet_main(const unsigned short* __restrict__ Abf,   // [B][D] bf16 bits
                  const unsigned short* __restrict__ Bbf,   // [B][D] bf16 bits (feat_oim)
                  const float* __restrict__ sqA,
                  const float* __restrict__ sqB,
                  const int*   __restrict__ ids,
                  float* __restrict__ pMax,   // [NCHUNK][B] partial row max(pos)
                  float* __restrict__ pMin,   // [NCHUNK][B] partial row min(neg)
                  int Bn, int Dk) {
    // dynamic LDS: 64 rows x Dk bf16, fragment-permuted:
    //   elem off = (((mblk*KB + kb)*32 + lane)*2 + chunk)*8 , mblk = row/16
    extern __shared__ __attribute__((aligned(32))) unsigned short As[];

    const int tid  = threadIdx.x;
    const int lane = tid & 31;
    const int wave = tid >> 5;
    const int mw = wave & 3;          // 0..3 : wave position along M
    const int nw = wave >> 2;         // 0..1 : wave position along N
    const int laneLo = lane & 15;
    const int laneHi = lane >> 4;
    const int KB = Dk >> 5;           // # of 32-wide K blocks
    const int rowBlock = blockIdx.x * 64;
    const int chunk    = blockIdx.y;
    const int nStart   = chunk * (Bn / NCHUNK);
    const int nSpan    = Bn / NCHUNK;

    // ---- cooperative A-tile load: coalesced global reads, permuted LDS stores
    {
        const int granPerRow = Dk >> 3;           // 16B granules per row
        const int totalGran  = 64 * granPerRow;
        for (int g = tid; g < totalGran; g += 256) {
            const int row = g / granPerRow;       // local row 0..63
            const int w   = g - row * granPerRow; // granule within row
            const int kb  = w >> 2;
            const int rem = w & 3;                // {0,8,16,24} = rem*8
            const int hi  = rem >> 1;
            const int chk = rem & 1;
            const int dlane = hi * 16 + (row & 15);
            const int dstElem =
                ((((row >> 4) * KB + kb) * 32 + dlane) * 2 + chk) * 8;
            v4u val = *(const v4u*)(Abf + (size_t)(rowBlock + row) * Dk + (size_t)w * 8);
            *(v4u*)(As + dstElem) = val;
        }
    }
    __syncthreads();

    // ---- per-wave row constants (C layout: lane<16 -> M=r, lane>=16 -> M=8+r)
    const int rowBase = rowBlock + mw * 16;
    float sA[8];
    int   iA[8];
    #pragma unroll
    for (int r = 0; r < 8; ++r) {
        const int i = rowBase + laneHi * 8 + r;
        sA[r] = sqA[i];
        iA[r] = ids[i];
    }

    float runMax[8], runMin[8];
    #pragma unroll
    for (int r = 0; r < 8; ++r) { runMax[r] = 0.0f; runMin[r] = FLT_MAX; }

    const unsigned short* AsWave = As + ((size_t)mw * KB * 32 + lane) * 16;

    // ---- N loop: each wave covers cols [colBase, colBase+64)
    for (int nBase = 0; nBase < nSpan; nBase += 128) {
        const int colBase = nStart + nBase + nw * 64;
        const int j0 = colBase + laneLo;
        const int j1 = j0 + 16;
        const int j2 = j0 + 32;
        const int j3 = j0 + 48;
        const v16bf* bp0 = (const v16bf*)(Bbf + (size_t)j0 * Dk + laneHi * 16);
        const v16bf* bp1 = (const v16bf*)(Bbf + (size_t)j1 * Dk + laneHi * 16);
        const v16bf* bp2 = (const v16bf*)(Bbf + (size_t)j2 * Dk + laneHi * 16);
        const v16bf* bp3 = (const v16bf*)(Bbf + (size_t)j3 * Dk + laneHi * 16);

        v8f c0 = {}, c1 = {}, c2 = {}, c3 = {};
        #pragma unroll 2
        for (int kb = 0; kb < KB; ++kb) {
            FragBits ab;
            const unsigned short* ap = AsWave + (size_t)kb * 32 * 16;
            ab.lo = *(const v4u*)(ap);
            ab.hi = *(const v4u*)(ap + 8);
            v16bf a  = __builtin_bit_cast(v16bf, ab);
            v16bf b0 = bp0[2 * kb];   // 32-elem (64B) stride per K block
            v16bf b1 = bp1[2 * kb];
            v16bf b2 = bp2[2 * kb];
            v16bf b3 = bp3[2 * kb];
            c0 = __builtin_amdgcn_wmma_f32_16x16x32_bf16(
                     false, a, false, b0, (short)0, c0, false, false);
            c1 = __builtin_amdgcn_wmma_f32_16x16x32_bf16(
                     false, a, false, b1, (short)0, c1, false, false);
            c2 = __builtin_amdgcn_wmma_f32_16x16x32_bf16(
                     false, a, false, b2, (short)0, c2, false, false);
            c3 = __builtin_amdgcn_wmma_f32_16x16x32_bf16(
                     false, a, false, b3, (short)0, c3, false, false);
        }

        // ---- fused epilogue: dist + masks + running row max/min
        const float sB0 = sqB[j0], sB1 = sqB[j1], sB2 = sqB[j2], sB3 = sqB[j3];
        const int   iB0 = ids[j0], iB1 = ids[j1], iB2 = ids[j2], iB3 = ids[j3];
        #pragma unroll
        for (int r = 0; r < 8; ++r) {
            const int i = rowBase + laneHi * 8 + r;
            const float g0 = c0[r], g1 = c1[r], g2 = c2[r], g3 = c3[r];
            {
                const float dist = sqrtf(fmaxf(sA[r] + sB0 - 2.0f * g0, 0.0f) + 1e-12f);
                const bool  same = (iA[r] == iB0);
                runMax[r] = fmaxf(runMax[r], (same && (i != j0)) ? dist : 0.0f);
                runMin[r] = fminf(runMin[r], dist + (same ? TL_BIG : 0.0f));
            }
            {
                const float dist = sqrtf(fmaxf(sA[r] + sB1 - 2.0f * g1, 0.0f) + 1e-12f);
                const bool  same = (iA[r] == iB1);
                runMax[r] = fmaxf(runMax[r], (same && (i != j1)) ? dist : 0.0f);
                runMin[r] = fminf(runMin[r], dist + (same ? TL_BIG : 0.0f));
            }
            {
                const float dist = sqrtf(fmaxf(sA[r] + sB2 - 2.0f * g2, 0.0f) + 1e-12f);
                const bool  same = (iA[r] == iB2);
                runMax[r] = fmaxf(runMax[r], (same && (i != j2)) ? dist : 0.0f);
                runMin[r] = fminf(runMin[r], dist + (same ? TL_BIG : 0.0f));
            }
            {
                const float dist = sqrtf(fmaxf(sA[r] + sB3 - 2.0f * g3, 0.0f) + 1e-12f);
                const bool  same = (iA[r] == iB3);
                runMax[r] = fmaxf(runMax[r], (same && (i != j3)) ? dist : 0.0f);
                runMin[r] = fminf(runMin[r], dist + (same ? TL_BIG : 0.0f));
            }
        }
    }

    // ---- reduce across the 16 lanes sharing each row (masks stay in-half)
    #pragma unroll
    for (int m = 1; m < 16; m <<= 1) {
        #pragma unroll
        for (int r = 0; r < 8; ++r) {
            runMax[r] = fmaxf(runMax[r], __shfl_xor(runMax[r], m, 32));
            runMin[r] = fminf(runMin[r], __shfl_xor(runMin[r], m, 32));
        }
    }

    // ---- combine the 2 N-waves via LDS (reuse A buffer after barrier)
    __syncthreads();                       // all waves done reading As
    float* redMax = (float*)As;            // [64][2]
    float* redMin = redMax + 64 * 2;       // [64][2]
    if (laneLo == 0) {
        #pragma unroll
        for (int r = 0; r < 8; ++r) {
            const int lrow = mw * 16 + laneHi * 8 + r;
            redMax[lrow * 2 + nw] = runMax[r];
            redMin[lrow * 2 + nw] = runMin[r];
        }
    }
    __syncthreads();
    if (tid < 64) {
        const float mx = fmaxf(redMax[tid * 2 + 0], redMax[tid * 2 + 1]);
        const float mn = fminf(redMin[tid * 2 + 0], redMin[tid * 2 + 1]);
        pMax[(size_t)chunk * Bn + rowBlock + tid] = mx;
        pMin[(size_t)chunk * Bn + rowBlock + tid] = mn;
    }
}

// ---- combine per-chunk partials, apply hinge --------------------------------
__global__ __launch_bounds__(256)
void finalize(const float* __restrict__ pMax, const float* __restrict__ pMin,
              float* __restrict__ out, int Bn) {
    const int row = blockIdx.x * 256 + threadIdx.x;
    if (row >= Bn) return;
    float mx = pMax[row];
    float mn = pMin[row];
    #pragma unroll
    for (int c = 1; c < NCHUNK; ++c) {
        mx = fmaxf(mx, pMax[(size_t)c * Bn + row]);
        mn = fminf(mn, pMin[(size_t)c * Bn + row]);
    }
    out[row] = fmaxf(mx - mn + TL_MARGIN, 0.0f);
}

// ---------------------------------------------------------------------------
extern "C" void kernel_launch(void* const* d_in, const int* in_sizes, int n_in,
                              void* d_out, int out_size, void* d_ws, size_t ws_size,
                              hipStream_t stream) {
    const float* feat = (const float*)d_in[0];
    const float* lut  = (const float*)d_in[1];
    const int*   id   = (const int*)d_in[2];

    const int Bn = in_sizes[2];            // 4096
    const int Dk = in_sizes[0] / Bn;       // 1024

    // workspace: [featBf 2*B*D][oimBf 2*B*D][sqA B][sqB B][pMax NCHUNK*B][pMin NCHUNK*B]
    char* ws = (char*)d_ws;
    unsigned short* featBf = (unsigned short*)ws;
    unsigned short* oimBf  = (unsigned short*)(ws + (size_t)Bn * Dk * 2);
    float* sqA  = (float*)(ws + (size_t)Bn * Dk * 4);
    float* sqB  = sqA + Bn;
    float* pMax = sqB + Bn;
    float* pMin = pMax + (size_t)NCHUNK * Bn;

    prep_feat<<<Bn, 256, 0, stream>>>(feat, featBf, sqA, Dk);
    prep_gather<<<Bn, 256, 0, stream>>>(lut, id, oimBf, sqB, Dk);

    const size_t ldsBytes = (size_t)64 * Dk * sizeof(unsigned short);  // 128 KB
    dim3 grid(Bn / 64, NCHUNK);
    triplet_main<<<grid, 256, ldsBytes, stream>>>(featBf, oimBf, sqA, sqB, id,
                                                  pMax, pMin, Bn, Dk);

    finalize<<<(Bn + 255) / 256, 256, 0, stream>>>(pMax, pMin, (float*)d_out, Bn);
}
